// MSDeformAttnTransformerEncoderOnly_70145405878415
// MI455X (gfx1250) — matrix operand order repaired and checked by reading, not compiled, not run on hardware
//
#include <hip/hip_runtime.h>
#include <cstdint>
#include <cstddef>

#define S_TOT 16660
#define BATCH 2
#define M_TOT (BATCH * S_TOT)   // 33320 tokens
#define NHD 8
#define DHD 32
#define DMODEL 256
#define DFFN 1024
#define NLAYER 6

typedef __attribute__((ext_vector_type(16))) _Float16 v16h;
typedef __attribute__((ext_vector_type(8)))  float    v8f;

// ---------------------------------------------------------------- WMMA core
__device__ __forceinline__ v8f wmma_f16f32(v16h a, v16h b, v8f c) {
  // D = A(16x32 f16) * B(32x16 f16) + C(16x16 f32)
  return __builtin_amdgcn_wmma_f32_16x16x32_f16(
      /*neg_a=*/false, a, /*neg_b=*/false, b,
      /*c_mod=*/(short)0, c, /*reuse_a=*/false, /*reuse_b=*/false);
}

// A fragment: LDS row-major [16][kstride] halves.
// Lane (kg*16+m): VGPR0-3 -> K = 8*kg + 0..7, VGPR4-7 -> K = 16 + 8*kg + 0..7
__device__ __forceinline__ v16h frag_a(const _Float16* sA, int kstride, int k0) {
  const int lane = threadIdx.x & 31;
  const int m  = lane & 15;
  const int kg = (lane >> 4) << 3;           // 0 or 8
  const _Float16* p = sA + m * kstride + k0;
  v16h a;
#pragma unroll
  for (int j = 0; j < 4; ++j) {
    a[2 * j]     = p[kg + 2 * j];
    a[2 * j + 1] = p[kg + 2 * j + 1];
  }
#pragma unroll
  for (int j = 0; j < 4; ++j) {
    a[8 + 2 * j]     = p[16 + kg + 2 * j];
    a[8 + 2 * j + 1] = p[16 + kg + 2 * j + 1];
  }
  return a;
}

// B fragment: LDS stores a 32x16 tile transposed as [n][k] (32 halves per n).
// Lane n+16*kg: elems 0-7 -> K = 8*kg + 0..7, elems 8-15 -> K = 16 + 8*kg + 0..7.
__device__ __forceinline__ v16h frag_b(const _Float16* sBtile) {
  const int lane = threadIdx.x & 31;
  const int n  = lane & 15;
  const int kg = (lane >> 4) << 3;           // 0 or 8
  const _Float16* p = sBtile + n * 32 + kg;
  v16h b;
#pragma unroll
  for (int j = 0; j < 16; ++j) b[j] = p[(j < 8) ? j : (8 + j)];
  return b;
}

// Vectorized A-tile stage: 16 rows x 256 halves, int4 (8-half) moves.
__device__ __forceinline__ void stage_a256(_Float16* sA, const _Float16* A,
                                           int m0, int M) {
  for (int v = threadIdx.x; v < 16 * 32; v += 256) {   // 32 int4 per row
    const int r = v >> 5;
    const int c8 = (v & 31) << 3;
    const int gm = m0 + r;
    if (gm < M)
      *(int4*)(sA + r * 256 + c8) = *(const int4*)(A + (size_t)gm * 256 + c8);
    else {
      const int4 z = {0, 0, 0, 0};
      *(int4*)(sA + r * 256 + c8) = z;
    }
  }
}

// ------------------------------------------------------- generic K=256 GEMM
// C[M,N] = A[M,256] * Wt^T + bias, with Wt pre-transposed as [N][256] f16.
template <int N, bool HOUT>
__global__ __launch_bounds__(256) void gemm_k256_kernel(
    const _Float16* __restrict__ A, const _Float16* __restrict__ Wt,
    const float* __restrict__ bias, void* __restrict__ Cout, int M) {
  constexpr int K = 256;
  constexpr int NTW = N / 128;               // N-tiles per wave
  __shared__ __align__(16) _Float16 sA[16 * K];
  __shared__ __align__(16) _Float16 sB[N * 32];
  const int m0 = blockIdx.x * 16;
  stage_a256(sA, A, m0, M);
  const int wave = threadIdx.x >> 5;
  v8f acc[NTW];
#pragma unroll
  for (int t = 0; t < NTW; ++t)
#pragma unroll
    for (int j = 0; j < 8; ++j) acc[t][j] = 0.0f;

  for (int k0 = 0; k0 < K; k0 += 32) {
    __syncthreads();
    for (int v = threadIdx.x; v < N * 4; v += 256) {   // 4 int4 per B row
      const int n = v >> 2;
      const int kk = (v & 3) << 3;
      *(int4*)(sB + n * 32 + kk) =
          *(const int4*)(Wt + (size_t)n * K + k0 + kk);
    }
    if (k0 + 32 < K && threadIdx.x < N)
      __builtin_prefetch(Wt + (size_t)threadIdx.x * K + k0 + 32, 0, 1);
    __syncthreads();
    const v16h a = frag_a(sA, K, k0);
#pragma unroll
    for (int t = 0; t < NTW; ++t) {
      const v16h b = frag_b(sB + (wave * NTW + t) * (16 * 32));
      acc[t] = wmma_f16f32(a, b, acc[t]);
    }
  }
  const int lane = threadIdx.x & 31;
  const int cn = lane & 15;
  const int rg = (lane >> 4) << 3;
#pragma unroll
  for (int t = 0; t < NTW; ++t) {
    const int n = (wave * NTW + t) * 16 + cn;
    const float bn = bias[n];
#pragma unroll
    for (int r = 0; r < 8; ++r) {
      const int row = m0 + rg + r;
      if (row < M) {
        const float v = acc[t][r] + bn;
        if constexpr (HOUT)
          ((_Float16*)Cout)[(size_t)row * N + n] = (_Float16)v;
        else
          ((float*)Cout)[(size_t)row * N + n] = v;
      }
    }
  }
}

// ------------------- fused offset+attn-weight projection: N = 256 + 128 = 384
__global__ __launch_bounds__(256) void gemm_qa_kernel(
    const _Float16* __restrict__ A, const _Float16* __restrict__ Wt,  // [384][256]
    const float* __restrict__ off_b, const float* __restrict__ aw_b,
    float* __restrict__ offb, float* __restrict__ awb, int M) {
  constexpr int K = 256, N = 384;
  __shared__ __align__(16) _Float16 sA[16 * K];
  __shared__ __align__(16) _Float16 sB[N * 32];
  const int m0 = blockIdx.x * 16;
  stage_a256(sA, A, m0, M);
  const int wave = threadIdx.x >> 5;
  v8f acc[3];
#pragma unroll
  for (int t = 0; t < 3; ++t)
#pragma unroll
    for (int j = 0; j < 8; ++j) acc[t][j] = 0.0f;

  for (int k0 = 0; k0 < K; k0 += 32) {
    __syncthreads();
    for (int v = threadIdx.x; v < N * 4; v += 256) {
      const int n = v >> 2;
      const int kk = (v & 3) << 3;
      *(int4*)(sB + n * 32 + kk) =
          *(const int4*)(Wt + (size_t)n * K + k0 + kk);
    }
    if (k0 + 32 < K && threadIdx.x < N)
      __builtin_prefetch(Wt + (size_t)threadIdx.x * K + k0 + 32, 0, 1);
    __syncthreads();
    const v16h a = frag_a(sA, K, k0);
#pragma unroll
    for (int t = 0; t < 3; ++t) {
      const v16h b = frag_b(sB + (wave * 3 + t) * (16 * 32));
      acc[t] = wmma_f16f32(a, b, acc[t]);
    }
  }
  const int lane = threadIdx.x & 31;
  const int cn = lane & 15;
  const int rg = (lane >> 4) << 3;
#pragma unroll
  for (int t = 0; t < 3; ++t) {
    const int n = (wave * 3 + t) * 16 + cn;   // tile index is wave-uniform
    const float bn = (n < 256) ? off_b[n] : aw_b[n - 256];
#pragma unroll
    for (int r = 0; r < 8; ++r) {
      const int row = m0 + rg + r;
      if (row < M) {
        const float v = acc[t][r] + bn;
        if (n < 256)
          offb[(size_t)row * 256 + n] = v;
        else
          awb[(size_t)row * 128 + (n - 256)] = v;
      }
    }
  }
}

// ---------------------------------------- out-projection + residual + LN (fused)
__global__ __launch_bounds__(256) void outproj_ln_kernel(
    const _Float16* __restrict__ A, const _Float16* __restrict__ Wt,  // [256][256]
    const float* __restrict__ bias, const float* __restrict__ ln_s,
    const float* __restrict__ ln_b, float* __restrict__ x,
    _Float16* __restrict__ xh, int M) {
  constexpr int K = 256, N = 256;
  __shared__ __align__(16) _Float16 sA[16 * K];
  __shared__ __align__(16) float sBC[16 * 256];  // union: f16 B stage / f32 C tile
  __shared__ float sMean[16], sRstd[16];
  _Float16* sB = (_Float16*)sBC;
  float* sC = sBC;

  const int m0 = blockIdx.x * 16;
  stage_a256(sA, A, m0, M);
  const int wave = threadIdx.x >> 5;
  v8f acc[2];
#pragma unroll
  for (int t = 0; t < 2; ++t)
#pragma unroll
    for (int j = 0; j < 8; ++j) acc[t][j] = 0.0f;

  for (int k0 = 0; k0 < K; k0 += 32) {
    __syncthreads();
    for (int v = threadIdx.x; v < N * 4; v += 256) {
      const int n = v >> 2;
      const int kk = (v & 3) << 3;
      *(int4*)(sB + n * 32 + kk) =
          *(const int4*)(Wt + (size_t)n * K + k0 + kk);
    }
    if (k0 + 32 < K)
      __builtin_prefetch(Wt + (size_t)threadIdx.x * K + k0 + 32, 0, 1);
    __syncthreads();
    const v16h a = frag_a(sA, K, k0);
#pragma unroll
    for (int t = 0; t < 2; ++t) {
      const v16h b = frag_b(sB + (wave * 2 + t) * 512);
      acc[t] = wmma_f16f32(a, b, acc[t]);
    }
  }
  __syncthreads();                            // sB dead -> reuse as sC
  const int lane = threadIdx.x & 31;
  const int cn = lane & 15;
  const int rg = (lane >> 4) << 3;
#pragma unroll
  for (int t = 0; t < 2; ++t) {
    const int n = (wave * 2 + t) * 16 + cn;
    const float bn = bias[n];
#pragma unroll
    for (int r = 0; r < 8; ++r) {
      const int row = m0 + rg + r;
      const float xo = (row < M) ? x[(size_t)row * 256 + n] : 0.0f;
      sC[(rg + r) * 256 + n] = xo + acc[t][r] + bn;
    }
  }
  __syncthreads();
  if (threadIdx.x < 16) {
    const int rr = threadIdx.x;
    float s1 = 0.0f, s2 = 0.0f;
    for (int c = 0; c < 256; ++c) {
      const float v = sC[rr * 256 + c];
      s1 += v; s2 += v * v;
    }
    const float mean = s1 * (1.0f / 256.0f);
    const float var = s2 * (1.0f / 256.0f) - mean * mean;
    sMean[rr] = mean;
    sRstd[rr] = rsqrtf(var + 1e-5f);
  }
  __syncthreads();
  for (int idx = threadIdx.x; idx < 16 * 256; idx += 256) {
    const int rr = idx >> 8, c = idx & 255;
    const int row = m0 + rr;
    if (row < M) {
      const float v = (sC[idx] - sMean[rr]) * sRstd[rr] * ln_s[c] + ln_b[c];
      x[(size_t)row * 256 + c] = v;
      xh[(size_t)row * 256 + c] = (_Float16)v;
    }
  }
}

// ------------------------------ fully fused FFN: relu(x@W1+b1)@W2+b2, +res, LN
__global__ __launch_bounds__(256) void ffn_ln_kernel(
    const _Float16* __restrict__ Xh,
    const _Float16* __restrict__ Wt1,   // [1024][256] (transposed ff1_w)
    const float* __restrict__ b1,
    const _Float16* __restrict__ Wt2,   // [256][1024] (transposed ff2_w)
    const float* __restrict__ b2, const float* __restrict__ ln_s,
    const float* __restrict__ ln_b, float* __restrict__ x, int M) {
  __shared__ __align__(16) _Float16 sA[16 * 256];   // 8 KB
  __shared__ __align__(16) float sBC[16 * 256];     // 16 KB (B stage / LN tile)
  __shared__ __align__(16) _Float16 sH[16 * DFFN];  // 32 KB hidden, LDS-resident
  __shared__ float sMean[16], sRstd[16];
  _Float16* sB = (_Float16*)sBC;
  float* sC = sBC;

  const int m0 = blockIdx.x * 16;
  stage_a256(sA, Xh, m0, M);
  const int wave = threadIdx.x >> 5;
  const int lane = threadIdx.x & 31;
  const int cn = lane & 15;
  const int rg = (lane >> 4) << 3;

  // ---- stage 1: h = relu(x @ W1 + b1), 4 column-blocks of 256
  for (int nb = 0; nb < 4; ++nb) {
    v8f acc[2];
#pragma unroll
    for (int t = 0; t < 2; ++t)
#pragma unroll
      for (int j = 0; j < 8; ++j) acc[t][j] = 0.0f;
    for (int k0 = 0; k0 < 256; k0 += 32) {
      __syncthreads();
      for (int v = threadIdx.x; v < 256 * 4; v += 256) {
        const int n = v >> 2;
        const int kk = (v & 3) << 3;
        *(int4*)(sB + n * 32 + kk) =
            *(const int4*)(Wt1 + (size_t)(nb * 256 + n) * 256 + k0 + kk);
      }
      if (k0 + 32 < 256)
        __builtin_prefetch(
            Wt1 + (size_t)(nb * 256 + threadIdx.x) * 256 + k0 + 32, 0, 1);
      __syncthreads();
      const v16h a = frag_a(sA, 256, k0);
#pragma unroll
      for (int t = 0; t < 2; ++t) {
        const v16h b = frag_b(sB + (wave * 2 + t) * 512);
        acc[t] = wmma_f16f32(a, b, acc[t]);
      }
    }
#pragma unroll
    for (int t = 0; t < 2; ++t) {
      const int ncol = nb * 256 + (wave * 2 + t) * 16 + cn;
      const float bn = b1[ncol];
#pragma unroll
      for (int r = 0; r < 8; ++r)
        sH[(rg + r) * DFFN + ncol] = (_Float16)fmaxf(acc[t][r] + bn, 0.0f);
    }
  }

  // ---- stage 2: y = h @ W2, K = 1024
  v8f acc2[2];
#pragma unroll
  for (int t = 0; t < 2; ++t)
#pragma unroll
    for (int j = 0; j < 8; ++j) acc2[t][j] = 0.0f;
  for (int k0 = 0; k0 < DFFN; k0 += 32) {
    __syncthreads();
    for (int v = threadIdx.x; v < 256 * 4; v += 256) {
      const int n = v >> 2;
      const int kk = (v & 3) << 3;
      *(int4*)(sB + n * 32 + kk) =
          *(const int4*)(Wt2 + (size_t)n * 1024 + k0 + kk);
    }
    if (k0 + 32 < DFFN)
      __builtin_prefetch(Wt2 + (size_t)threadIdx.x * 1024 + k0 + 32, 0, 1);
    __syncthreads();
    const v16h a = frag_a(sH, DFFN, k0);
#pragma unroll
    for (int t = 0; t < 2; ++t) {
      const v16h b = frag_b(sB + (wave * 2 + t) * 512);
      acc2[t] = wmma_f16f32(a, b, acc2[t]);
    }
  }
  __syncthreads();                            // sB dead -> reuse as sC
#pragma unroll
  for (int t = 0; t < 2; ++t) {
    const int n = (wave * 2 + t) * 16 + cn;
    const float bn = b2[n];
#pragma unroll
    for (int r = 0; r < 8; ++r) {
      const int row = m0 + rg + r;
      const float xo = (row < M) ? x[(size_t)row * 256 + n] : 0.0f;
      sC[(rg + r) * 256 + n] = xo + acc2[t][r] + bn;
    }
  }
  __syncthreads();
  if (threadIdx.x < 16) {
    const int rr = threadIdx.x;
    float s1 = 0.0f, s2 = 0.0f;
    for (int c = 0; c < 256; ++c) {
      const float v = sC[rr * 256 + c];
      s1 += v; s2 += v * v;
    }
    const float mean = s1 * (1.0f / 256.0f);
    const float var = s2 * (1.0f / 256.0f) - mean * mean;
    sMean[rr] = mean;
    sRstd[rr] = rsqrtf(var + 1e-5f);
  }
  __syncthreads();
  for (int idx = threadIdx.x; idx < 16 * 256; idx += 256) {
    const int rr = idx >> 8, c = idx & 255;
    const int row = m0 + rr;
    if (row < M)
      x[(size_t)row * 256 + c] =
          (sC[idx] - sMean[rr]) * sRstd[rr] * ln_s[c] + ln_b[c];
  }
}

// ------------------------------------------------ deformable attention gather
__device__ __forceinline__ float dtap(const _Float16* vb, int xi, int yi, int Wl) {
  const bool valid = (xi >= 0) && (xi < Wl) && (yi >= 0) && (yi < Wl);
  const int cx = min(max(xi, 0), Wl - 1);
  const int cy = min(max(yi, 0), Wl - 1);
  const float v = (float)vb[(size_t)(cy * Wl + cx) * (NHD * DHD)];
  return valid ? v : 0.0f;
}

__global__ __launch_bounds__(256) void deform_kernel(
    const _Float16* __restrict__ valh, const float* __restrict__ off,
    const float* __restrict__ aw, _Float16* __restrict__ attnh) {
  const int wave = threadIdx.x >> 5;
  const int lane = threadIdx.x & 31;           // lane = channel within head
  const int gid = blockIdx.x * 8 + wave;       // (b, s, h)
  if (gid >= M_TOT * NHD) return;
  const int b = gid / (S_TOT * NHD);
  const int rs = gid - b * (S_TOT * NHD);
  const int s = rs / NHD;
  const int h = rs - s * NHD;

  const int DIMS[4] = {112, 56, 28, 14};
  const int LST[4]  = {0, 12544, 15680, 16464};
  int l0, hw;
  if (s < 12544)      { l0 = 0; hw = s; }
  else if (s < 15680) { l0 = 1; hw = s - 12544; }
  else if (s < 16464) { l0 = 2; hw = s - 15680; }
  else                { l0 = 3; hw = s - 16464; }
  const int W0 = DIMS[l0];
  const float refx = ((float)(hw % W0) + 0.5f) / (float)W0;
  const float refy = ((float)(hw / W0) + 0.5f) / (float)W0;

  const float* offp = off + (size_t)(b * S_TOT + s) * 256 + h * 32;
  const float* awp  = aw  + (size_t)(b * S_TOT + s) * 128 + h * 16;

  float acc = 0.0f;
#pragma unroll
  for (int l = 0; l < 4; ++l) {
    const int Wl = DIMS[l];
    const float fW = (float)Wl;
    const _Float16* vb =
        valh + ((size_t)(b * S_TOT + LST[l]) * NHD + h) * DHD + lane;
#pragma unroll
    for (int p = 0; p < 4; ++p) {
      const float a = awp[l * 4 + p];
      // loc = ref + off/W ; grid_sample align_corners=False
      const float px = (refx + offp[l * 8 + p * 2 + 0] / fW) * fW - 0.5f;
      const float py = (refy + offp[l * 8 + p * 2 + 1] / fW) * fW - 0.5f;
      const float fx = floorf(px), fy = floorf(py);
      const int x0 = (int)fx, y0 = (int)fy;
      const float lx = px - fx, ly = py - fy;
      const float v00 = dtap(vb, x0,     y0,     Wl);
      const float v10 = dtap(vb, x0 + 1, y0,     Wl);
      const float v01 = dtap(vb, x0,     y0 + 1, Wl);
      const float v11 = dtap(vb, x0 + 1, y0 + 1, Wl);
      acc += a * ((1.0f - lx) * (1.0f - ly) * v00 + lx * (1.0f - ly) * v10 +
                  (1.0f - lx) * ly * v01 + lx * ly * v11);
    }
  }
  attnh[(size_t)(b * S_TOT + s) * 256 + h * DHD + lane] = (_Float16)acc;
}

// -------------------------------------------------------------- small kernels
__global__ void softmax16_kernel(float* __restrict__ aw, int total) {
  const int i = blockIdx.x * blockDim.x + threadIdx.x;
  if (i >= total) return;
  float* p = aw + (size_t)i * 16;
  float mx = p[0];
#pragma unroll
  for (int j = 1; j < 16; ++j) mx = fmaxf(mx, p[j]);
  float e[16], sum = 0.0f;
#pragma unroll
  for (int j = 0; j < 16; ++j) { e[j] = __expf(p[j] - mx); sum += e[j]; }
  const float inv = 1.0f / sum;
#pragma unroll
  for (int j = 0; j < 16; ++j) p[j] = e[j] * inv;
}

__global__ __launch_bounds__(256) void pack_kernel(
    const float* __restrict__ s0, const float* __restrict__ s1,
    const float* __restrict__ s2, const float* __restrict__ s3,
    const float* __restrict__ p0, const float* __restrict__ p1,
    const float* __restrict__ p2, const float* __restrict__ p3,
    const float* __restrict__ lvl, float* __restrict__ x,
    float* __restrict__ pos) {
  const int bs = blockIdx.x;
  const int d = threadIdx.x;
  const int b = bs / S_TOT;
  const int s = bs - b * S_TOT;
  const float* sp; const float* pp; int HW, l, hw;
  if (s < 12544)      { l = 0; sp = s0; pp = p0; HW = 12544; hw = s; }
  else if (s < 15680) { l = 1; sp = s1; pp = p1; HW = 3136;  hw = s - 12544; }
  else if (s < 16464) { l = 2; sp = s2; pp = p2; HW = 784;   hw = s - 15680; }
  else                { l = 3; sp = s3; pp = p3; HW = 196;   hw = s - 16464; }
  const size_t si = ((size_t)b * 256 + d) * HW + hw;
  x[(size_t)bs * 256 + d] = sp[si];
  pos[(size_t)bs * 256 + d] = pp[si] + lvl[l * 256 + d];
}

__global__ void make_qx_kernel(const float* __restrict__ x,
                               const float* __restrict__ pos,
                               _Float16* __restrict__ qh,
                               _Float16* __restrict__ xh, size_t n) {
  const size_t i = (size_t)blockIdx.x * blockDim.x + threadIdx.x;
  if (i >= n) return;
  const float xv = x[i];
  qh[i] = (_Float16)(xv + pos[i]);
  xh[i] = (_Float16)xv;
}

// f32 [NL][K][N] -> f16 transposed [NL][rowstride][K] (rows row0..row0+N-1)
__global__ void transpose_f16_kernel(const float* __restrict__ in,
                                     _Float16* __restrict__ out, int K, int N,
                                     int rowstride, int row0, size_t total) {
  const size_t i = (size_t)blockIdx.x * blockDim.x + threadIdx.x;
  if (i >= total) return;
  const int per = K * N;
  const int l = (int)(i / per);
  const int rem = (int)(i % per);
  const int n = rem / K;
  const int k = rem - n * K;
  out[((size_t)l * rowstride + row0 + n) * K + k] =
      (_Float16)in[((size_t)l * K + k) * N + n];
}

__global__ void finalize_kernel(const float* __restrict__ x,
                                float* __restrict__ out, size_t n) {
  const size_t i = (size_t)blockIdx.x * blockDim.x + threadIdx.x;
  if (i < n) out[i] = x[i];
  if (i < 12) {
    const int tail[12] = {112, 112, 56, 56, 28, 28, 14, 14,
                          0, 12544, 15680, 16464};
    ((int*)(out + n))[i] = tail[i];
  }
}

// ------------------------------------------------------------------- launcher
extern "C" void kernel_launch(void* const* d_in, const int* in_sizes, int n_in,
                              void* d_out, int out_size, void* d_ws,
                              size_t ws_size, hipStream_t stream) {
  (void)in_sizes; (void)n_in; (void)out_size; (void)ws_size;
  const float* src[4]  = {(const float*)d_in[0], (const float*)d_in[2],
                          (const float*)d_in[4], (const float*)d_in[6]};
  const float* posi[4] = {(const float*)d_in[1], (const float*)d_in[3],
                          (const float*)d_in[5], (const float*)d_in[7]};
  const float* lvl   = (const float*)d_in[8];
  const float* off_w = (const float*)d_in[9];
  const float* off_b = (const float*)d_in[10];
  const float* aw_w  = (const float*)d_in[11];
  const float* aw_b  = (const float*)d_in[12];
  const float* val_w = (const float*)d_in[13];
  const float* val_b = (const float*)d_in[14];
  const float* out_w = (const float*)d_in[15];
  const float* out_b = (const float*)d_in[16];
  const float* ln1_s = (const float*)d_in[17];
  const float* ln1_b = (const float*)d_in[18];
  const float* ff1_w = (const float*)d_in[19];
  const float* ff1_b = (const float*)d_in[20];
  const float* ff2_w = (const float*)d_in[21];
  const float* ff2_b = (const float*)d_in[22];
  const float* ln2_s = (const float*)d_in[23];
  const float* ln2_b = (const float*)d_in[24];

  const int M = M_TOT;
  const size_t nElem = (size_t)M * 256;

  char* w = (char*)d_ws;
  auto take = [&](size_t bytes) -> void* {
    void* p = (void*)w;
    w += (bytes + 255) & ~(size_t)255;
    return p;
  };
  float*    x     = (float*)take(nElem * 4);
  float*    pos   = (float*)take(nElem * 4);
  _Float16* qh    = (_Float16*)take(nElem * 2);
  _Float16* xh    = (_Float16*)take(nElem * 2);
  float*    offb  = (float*)take(nElem * 4);
  float*    awb   = (float*)take((size_t)M * 128 * 4);
  _Float16* valh  = (_Float16*)take(nElem * 2);
  _Float16* attnh = (_Float16*)take(nElem * 2);
  _Float16* Wtqa  = (_Float16*)take((size_t)NLAYER * 384 * 256 * 2);
  _Float16* Wtval = (_Float16*)take((size_t)NLAYER * 256 * 256 * 2);
  _Float16* Wtout = (_Float16*)take((size_t)NLAYER * 256 * 256 * 2);
  _Float16* Wtff1 = (_Float16*)take((size_t)NLAYER * 1024 * 256 * 2);
  _Float16* Wtff2 = (_Float16*)take((size_t)NLAYER * 256 * 1024 * 2);

  const int MT = (M + 15) / 16;

  // one-time f32 -> f16 transposed weight copies ([N][K] row-major, per layer)
  {
    size_t n;
    n = (size_t)NLAYER * 256 * 256;
    transpose_f16_kernel<<<(int)((n + 255) / 256), 256, 0, stream>>>(
        off_w, Wtqa, 256, 256, 384, 0, n);
    n = (size_t)NLAYER * 256 * 128;
    transpose_f16_kernel<<<(int)((n + 255) / 256), 256, 0, stream>>>(
        aw_w, Wtqa, 256, 128, 384, 256, n);
    n = (size_t)NLAYER * 256 * 256;
    transpose_f16_kernel<<<(int)((n + 255) / 256), 256, 0, stream>>>(
        val_w, Wtval, 256, 256, 256, 0, n);
    transpose_f16_kernel<<<(int)((n + 255) / 256), 256, 0, stream>>>(
        out_w, Wtout, 256, 256, 256, 0, n);
    n = (size_t)NLAYER * 256 * 1024;
    transpose_f16_kernel<<<(int)((n + 255) / 256), 256, 0, stream>>>(
        ff1_w, Wtff1, 256, 1024, 1024, 0, n);
    transpose_f16_kernel<<<(int)((n + 255) / 256), 256, 0, stream>>>(
        ff2_w, Wtff2, 1024, 256, 256, 0, n);
  }

  pack_kernel<<<M, 256, 0, stream>>>(src[0], src[1], src[2], src[3], posi[0],
                                     posi[1], posi[2], posi[3], lvl, x, pos);

  for (int i = 0; i < NLAYER; ++i) {
    make_qx_kernel<<<(int)(nElem / 256), 256, 0, stream>>>(x, pos, qh, xh, nElem);
    gemm_qa_kernel<<<MT, 256, 0, stream>>>(
        qh, Wtqa + (size_t)i * 384 * 256, off_b + i * 256, aw_b + i * 128,
        offb, awb, M);
    gemm_k256_kernel<256, true><<<MT, 256, 0, stream>>>(
        xh, Wtval + (size_t)i * 256 * 256, val_b + i * 256, valh, M);
    softmax16_kernel<<<(M * 8 + 255) / 256, 256, 0, stream>>>(awb, M * 8);
    deform_kernel<<<M, 256, 0, stream>>>(valh, offb, awb, attnh);
    outproj_ln_kernel<<<MT, 256, 0, stream>>>(
        attnh, Wtout + (size_t)i * 256 * 256, out_b + i * 256, ln1_s + i * 256,
        ln1_b + i * 256, x, xh, M);
    ffn_ln_kernel<<<MT, 256, 0, stream>>>(
        xh, Wtff1 + (size_t)i * 1024 * 256, ff1_b + i * 1024,
        Wtff2 + (size_t)i * 256 * 1024, ff2_b + i * 256, ln2_s + i * 256,
        ln2_b + i * 256, x, M);
  }

  finalize_kernel<<<(int)((nElem + 255) / 256), 256, 0, stream>>>(
      x, (float*)d_out, nElem);
}
